// QuantDenseBNBlock_84585085927956
// MI455X (gfx1250) — 8-bit matrix operand order repaired and checked by reading, compiled, not run on hardware
//
#include <hip/hip_runtime.h>
#include <hip/hip_bf16.h>

typedef __attribute__((ext_vector_type(16))) int   v16i;
typedef __attribute__((ext_vector_type(8)))  float v8f;

#define B_ROWS   32768
#define D_IN     2048
#define D_H      512
#define KSTEP    128
#define MBLK     32
#define LDSTRIDE 144
#define ABUF     (MBLK * LDSTRIDE)

#define FP8_P1 0x38u
#define FP8_M1 0xB8u

#if __has_builtin(__builtin_amdgcn_sched_barrier)
#define SCHED_BARRIER() __builtin_amdgcn_sched_barrier(0)
#else
#define SCHED_BARRIER()
#endif

#define WMMA_FP8(A, B, C) \
    __builtin_amdgcn_wmma_f32_16x16x128_fp8_fp8((A), (B), (short)0, (C), false, false)

__device__ __forceinline__ unsigned int pack4sign(float4 f) {
    unsigned int u0 = __float_as_uint(f.x);
    unsigned int u1 = __float_as_uint(f.y);
    unsigned int u2 = __float_as_uint(f.z);
    unsigned int u3 = __float_as_uint(f.w);
    unsigned int lo = __builtin_amdgcn_perm(u1, u0, 0x0C0C0703u);
    unsigned int hi = __builtin_amdgcn_perm(u3, u2, 0x07030C0Cu);
    return ((lo | hi) & 0x80808080u) | 0x38383838u;
}

__device__ __forceinline__ v16i loadBfrag(const unsigned char* wb) {
    v16i bf;
    #pragma unroll
    for (int c = 0; c < 4; ++c) {
        int4 d = *(const int4*)(wb + 32 * c);
        bf[4 * c]     = d.x;
        bf[4 * c + 1] = d.y;
        bf[4 * c + 2] = d.z;
        bf[4 * c + 3] = d.w;
    }
    return bf;
}

__global__ void qdense_prep_kernel(const float* __restrict__ W,
                                   const float* __restrict__ b,
                                   const float* __restrict__ beta,
                                   const float* __restrict__ mean,
                                   const float* __restrict__ var,
                                   unsigned char* __restrict__ Wq,
                                   float* __restrict__ colScale,
                                   float* __restrict__ colBias) {
    int tid = blockIdx.x * blockDim.x + threadIdx.x;
    const int total = D_H * (D_IN / 4);
    if (tid < total) {
        int n  = tid / (D_IN / 4);
        int k4 = (tid % (D_IN / 4)) * 4;
        unsigned int packed = 0;
        #pragma unroll
        for (int j = 0; j < 4; ++j) {
            float w = W[(size_t)(k4 + j) * D_H + n];
            packed |= ((w >= 0.0f) ? FP8_P1 : FP8_M1) << (8 * j);
        }
        *(unsigned int*)(Wq + (size_t)n * D_IN + k4) = packed;
    }
    if (tid < D_H) {
        float inv = rsqrtf(var[tid] + 1e-3f);
        colScale[tid] = inv;
        colBias[tid]  = (b[tid] - mean[tid]) * inv + beta[tid];
    }
}

__global__ __launch_bounds__(256)
void qdense_bgemm_kernel(const float* __restrict__ X,
                         const unsigned char* __restrict__ Wq,
                         const float* __restrict__ colScale,
                         const float* __restrict__ colBias,
                         float* __restrict__ out) {
    __shared__ unsigned char ldsA[2 * ABUF];

    const int tid  = threadIdx.x;
    const int wave = tid >> 5;
    const int lane = tid & 31;
    const int m0   = blockIdx.x * MBLK;

    const int mw    = wave >> 2;
    const int nw    = wave & 3;
    const int n0    = nw * 128;
    const int mbase = mw * 16;

    v8f acc[8];
    #pragma unroll
    for (int t = 0; t < 8; ++t)
        #pragma unroll
        for (int r = 0; r < 8; ++r) acc[t][r] = 0.0f;

    const int arow = lane & 15;
    const int aoff = (lane >> 4) * 16;

    const int bcol = lane & 15;
    const int boff = (lane >> 4) * 16;

    const int xrow  = tid >> 3;
    const int xkseg = (tid & 7) * 16;

    const float* xbase = X + (size_t)(m0 + xrow) * D_IN + xkseg;
    const int ldstOff = xrow * LDSTRIDE + xkseg;
    const int abOff   = (mbase + arow) * LDSTRIDE + aoff;

    float4 xf[4];
    #pragma unroll
    for (int j = 0; j < 4; ++j) xf[j] = *(const float4*)(xbase + 4 * j);

    int buf = 0;
    for (int kt = 0; kt < D_IN; kt += KSTEP) {
        unsigned int* ldst = (unsigned int*)(ldsA + buf * ABUF + ldstOff);
        #pragma unroll
        for (int j = 0; j < 4; ++j) ldst[j] = pack4sign(xf[j]);
        __syncthreads();

        const unsigned char* ab = ldsA + buf * ABUF + abOff;
        v16i a;
        #pragma unroll
        for (int c = 0; c < 8; ++c) {
            uint2 d = *(const uint2*)(ab + 32 * (c >> 1) + 8 * (c & 1));
            a[2 * c]     = (int)d.x;
            a[2 * c + 1] = (int)d.y;
        }
        buf ^= 1;

        const unsigned char* wbase = Wq + (size_t)(n0 + bcol) * D_IN + kt + boff;
        const int knext = (kt + KSTEP) & (D_IN - 1);

        v16i b0 = loadBfrag(wbase);
        v16i b1 = loadBfrag(wbase + 1 * 16 * D_IN);
        SCHED_BARRIER();
        acc[0] = WMMA_FP8(a, b0, acc[0]);
        b0 = loadBfrag(wbase + 2 * 16 * D_IN);
        SCHED_BARRIER();
        acc[1] = WMMA_FP8(a, b1, acc[1]);
        b1 = loadBfrag(wbase + 3 * 16 * D_IN);
        SCHED_BARRIER();
        acc[2] = WMMA_FP8(a, b0, acc[2]);
        b0 = loadBfrag(wbase + 4 * 16 * D_IN);
        #pragma unroll
        for (int j = 0; j < 4; ++j)
            xf[j] = *(const float4*)(xbase + knext + 4 * j);
        SCHED_BARRIER();
        acc[3] = WMMA_FP8(a, b1, acc[3]);
        b1 = loadBfrag(wbase + 5 * 16 * D_IN);
        SCHED_BARRIER();
        acc[4] = WMMA_FP8(a, b0, acc[4]);
        b0 = loadBfrag(wbase + 6 * 16 * D_IN);
        SCHED_BARRIER();
        acc[5] = WMMA_FP8(a, b1, acc[5]);
        b1 = loadBfrag(wbase + 7 * 16 * D_IN);
        SCHED_BARRIER();
        acc[6] = WMMA_FP8(a, b0, acc[6]);
        SCHED_BARRIER();
        acc[7] = WMMA_FP8(a, b1, acc[7]);
    }

    #pragma unroll
    for (int t = 0; t < 8; ++t) {
        int n = n0 + t * 16 + (lane & 15);
        float sc = colScale[n];
        float bs = colBias[n];
        int rbase = m0 + mbase + (lane >> 4) * 8;
        #pragma unroll
        for (int r = 0; r < 8; ++r) {
            out[(size_t)(rbase + r) * D_H + n] = acc[t][r] * sc + bs;
        }
    }
}

extern "C" void kernel_launch(void* const* d_in, const int* in_sizes, int n_in,
                              void* d_out, int out_size, void* d_ws, size_t ws_size,
                              hipStream_t stream) {
    const float* X    = (const float*)d_in[0];
    const float* W    = (const float*)d_in[1];
    const float* b    = (const float*)d_in[2];
    const float* beta = (const float*)d_in[3];
    const float* mean = (const float*)d_in[4];
    const float* var  = (const float*)d_in[5];
    float* out = (float*)d_out;

    unsigned char* Wq = (unsigned char*)d_ws;
    float* colScale = (float*)((char*)d_ws + (size_t)D_H * D_IN);
    float* colBias  = colScale + D_H;

    const int prepThreads = D_H * (D_IN / 4);
    qdense_prep_kernel<<<(prepThreads + 255) / 256, 256, 0, stream>>>(
        W, b, beta, mean, var, Wq, colScale, colBias);

    qdense_bgemm_kernel<<<B_ROWS / MBLK, 256, 0, stream>>>(
        X, Wq, colScale, colBias, out);
}
